// LlamaQAttention2_54365696033093
// MI455X (gfx1250) — matrix operand order repaired and checked by reading, not compiled, not run on hardware
//
#include <hip/hip_runtime.h>
#include <hip/hip_bf16.h>

// ---------------- problem constants ----------------
constexpr int B  = 2;
constexpr int S  = 1024;
constexpr int H  = 4096;
constexpr int NH = 32;
constexpr int HD = 128;
constexpr int G  = 128;
constexpr int OQKV = 3 * H;           // 12288
constexpr int MROWS = B * S;          // 2048

typedef __attribute__((ext_vector_type(16))) _Float16 v16h;
typedef __attribute__((ext_vector_type(8)))  float    v8f;

union F16x16 { v16h h; uint4 u[2]; };

__device__ __forceinline__ v8f zero8() {
  v8f z;
  #pragma unroll
  for (int i = 0; i < 8; ++i) z[i] = 0.0f;
  return z;
}

// ---- CDNA5 async global->LDS 16B copy (per lane), tracked by ASYNCcnt ----
// LDS byte offset is addr[31:0] of the generic shared-aperture address.
__device__ __forceinline__ void copy16_async(const _Float16* g, _Float16* l) {
  unsigned lds = (unsigned)(size_t)l;
  asm volatile("global_load_async_to_lds_b128 %0, %1, off"
               :: "v"(lds), "v"(g) : "memory");
}
__device__ __forceinline__ void async_join() {
  asm volatile("s_wait_asynccnt 0" ::: "memory");
}

// ---------------- kernel 1: f32 -> f16 convert ----------------
__global__ void convert_f32_to_f16(const float* __restrict__ in,
                                   _Float16* __restrict__ out, int n) {
  int i = blockIdx.x * blockDim.x + threadIdx.x;
  if (i < n) out[i] = (_Float16)in[i];
}

// ---------------- kernel 2: AWQ int4 dequant -> transposed f16 Wt[N][K] ----
// qw: [rows(K), cols_packed] int32; qz: [rows/G, cols_packed]; scales: [rows/G, ncols]
// Block handles a 64(k) x 32(n) tile; LDS transpose so global writes are 16B chunks.
__global__ __launch_bounds__(256)
void dequant_awq_T(const int* __restrict__ qw,
                   const float* __restrict__ scales,
                   const int* __restrict__ qz,
                   _Float16* __restrict__ outT,   // [ncols, rows]
                   int rows, int ncols) {
  __shared__ _Float16 T[32][72];                 // +8 pad, 144B row stride (16B mult)
  const int cols_packed = ncols >> 3;
  const int tiles_n = ncols >> 5;
  const int tn = blockIdx.x % tiles_n;
  const int tr = blockIdx.x / tiles_n;
  const int r0 = tr * 64, n0 = tn * 32;

  // phase 1: dequant 64x32 tile into LDS (transposed)
  {
    int rl = threadIdx.x >> 2;                   // 0..63
    int cl = threadIdx.x & 3;                    // packed col 0..3
    int r = r0 + rl;
    int c = (n0 >> 3) + cl;
    int grp = r / G;
    unsigned w = (unsigned)qw[(size_t)r * cols_packed + c];
    unsigned z = (unsigned)qz[(size_t)grp * cols_packed + c];
    const int INV[8] = {0, 4, 1, 5, 2, 6, 3, 7}; // argsort of AWQ order
    #pragma unroll
    for (int i = 0; i < 8; ++i) {
      int sh = INV[i] * 4;
      float wi = (float)((w >> sh) & 15u);
      float zi = (float)((z >> sh) & 15u);
      float sc = scales[(size_t)grp * ncols + c * 8 + i];
      T[cl * 8 + i][rl] = (_Float16)((wi - zi) * sc);
    }
  }
  __syncthreads();
  // phase 2: coalesced 16B writes of Wt rows
  {
    int nl = threadIdx.x >> 3;                   // 0..31
    int kc = (threadIdx.x & 7) * 8;              // 0..56
    uint4 d = *(const uint4*)&T[nl][kc];
    *(uint4*)&outT[(size_t)(n0 + nl) * rows + r0 + kc] = d;
  }
}

// ---------------- kernel 3/7: tiled WMMA GEMM (B pre-transposed) ----------
// C[M,N] = A[M,K] * W[K,N] with W given as Wt[N,K] row-major.
// 128x128 tile, TK=32, 256 threads = 8 waves; async LDS staging.
template <bool OUT_F32>
__global__ __launch_bounds__(256)
void gemm_wmma(const _Float16* __restrict__ A, const _Float16* __restrict__ Bt_g,
               void* __restrict__ Cout, int M, int N, int K) {
  __shared__ _Float16 Ash[128][32];   // A tile: [m][k]
  __shared__ _Float16 Bsh[128][32];   // Wt tile: [n][k]
  const int tid  = threadIdx.x;
  const int wave = tid >> 5;
  const int lane = tid & 31;
  const int lhalf = lane & 15, lhi = lane >> 4;
  const int m0 = blockIdx.y * 128;
  const int n0 = blockIdx.x * 128;

  v8f acc[8];
  #pragma unroll
  for (int t = 0; t < 8; ++t) acc[t] = zero8();

  for (int k0 = 0; k0 < K; k0 += 32) {
    #pragma unroll
    for (int rep = 0; rep < 2; ++rep) {
      int cid = tid + rep * 256;
      int r = cid >> 2;                // 0..127
      int c = (cid & 3) * 8;           // 0,8,16,24
      copy16_async(A    + (size_t)(m0 + r) * K + k0 + c, &Ash[r][c]);
      copy16_async(Bt_g + (size_t)(n0 + r) * K + k0 + c, &Bsh[r][c]);
    }
    async_join();
    __syncthreads();

    // A fragment: 16-bit A 16x32 (lanes 0-15: K 0..7 & 16..23; 16-31: K 8..15 & 24..31)
    F16x16 af;
    int am  = wave * 16 + lhalf;
    int akb = lhi * 8;
    af.u[0] = *(const uint4*)&Ash[am][akb];
    af.u[1] = *(const uint4*)&Ash[am][akb + 16];

    #pragma unroll
    for (int t = 0; t < 8; ++t) {
      // B fragment: 32x16 (lanes 0-15: K 0..7 & 16..23 at N=lane; 16-31: K 8..15 & 24..31)
      F16x16 bf;
      int bn  = t * 16 + lhalf;
      int bkb = lhi * 8;
      bf.u[0] = *(const uint4*)&Bsh[bn][bkb];
      bf.u[1] = *(const uint4*)&Bsh[bn][bkb + 16];
      acc[t] = __builtin_amdgcn_wmma_f32_16x16x32_f16(
          false, af.h, false, bf.h, (short)0, acc[t], false, false);
    }
    __syncthreads();
  }

  #pragma unroll
  for (int t = 0; t < 8; ++t) {
    #pragma unroll
    for (int v = 0; v < 8; ++v) {
      size_t m = (size_t)(m0 + wave * 16 + v + 8 * lhi);
      size_t n = (size_t)(n0 + t * 16 + lhalf);
      if constexpr (OUT_F32) ((float*)Cout)[m * N + n] = acc[t][v];
      else ((_Float16*)Cout)[m * N + n] = (_Float16)acc[t][v];
    }
  }
}

// ---------------- kernel 4: in-place RoPE on q,k of qkv buffer -------------
__global__ void rope_kernel(const int* __restrict__ positions,
                            _Float16* __restrict__ qkv, int n) {
  int idx = blockIdx.x * blockDim.x + threadIdx.x;
  if (idx >= n) return;
  const int half = HD / 2;               // 64
  int i    = idx & (half - 1);
  int rest = idx >> 6;
  int head = rest & (NH - 1); rest >>= 5;
  int qk   = rest & 1;        rest >>= 1;
  int s    = rest & (S - 1);
  int b    = rest >> 10;
  float pos  = (float)positions[s];
  float invf = __expf(-(2.0f * (float)i / (float)HD) * 9.210340371976184f); // ln(1e4)
  float fr = pos * invf;
  float c = __cosf(fr), sn = __sinf(fr);
  _Float16* p = qkv + (size_t)(b * S + s) * OQKV + qk * H + head * HD;
  float x1 = (float)p[i];
  float x2 = (float)p[i + half];
  p[i]        = (_Float16)(x1 * c - x2 * sn);
  p[i + half] = (_Float16)(x1 * sn + x2 * c);
}

// ---------------- kernel 5: V transpose: qkv v-part -> Vt[B,NH,HD,S] -------
__global__ void transpose_v(const _Float16* __restrict__ qkv,
                            _Float16* __restrict__ vt, int n) {
  int idx = blockIdx.x * blockDim.x + threadIdx.x;
  if (idx >= n) return;
  int s    = idx & (S - 1);
  int rest = idx >> 10;
  int hd   = rest & (HD - 1); rest >>= 7;
  int h    = rest & (NH - 1);
  int b    = rest >> 5;
  vt[idx] = qkv[(size_t)(b * S + s) * OQKV + 2 * H + h * HD + hd];
}

// ---------------- kernel 6: causal flash attention (WMMA) ------------------
// qkv: [B*S, 3H] f16 (q/k roped); vt: [B,NH,HD,S]; out: [B*S, H] f16.
__global__ __launch_bounds__(256)
void attn_kernel(const _Float16* __restrict__ qkv,
                 const _Float16* __restrict__ vt,
                 _Float16* __restrict__ out) {
  __shared__ _Float16 Ksh[32][128];      // 32 keys x 128 hd
  __shared__ _Float16 Vt[128][32];       // Vt chunk: [hd][key]
  __shared__ _Float16 Psh[8][16][32];    // wave-private P staging
  const int tid = threadIdx.x, wave = tid >> 5, lane = tid & 31;
  const int lhalf = lane & 15, lhi = lane >> 4;
  const int q0 = blockIdx.x * 128;
  const int bh = blockIdx.y;
  const int head = bh % NH, b = bh / NH;
  const size_t rs = OQKV;
  const _Float16* qb  = qkv + (size_t)b * S * rs + head * HD;
  const _Float16* kb_ = qb + H;
  const _Float16* vtb = vt + ((size_t)(b * NH + head)) * HD * S;
  const float scale = 0.08838834764831845f;  // 1/sqrt(128)

  // Q fragments for K-dim = 128 (4 x K=32)
  F16x16 qf[4];
  {
    int qrow = q0 + wave * 16 + lhalf;
    #pragma unroll
    for (int c = 0; c < 4; ++c) {
      int hd0 = c * 32 + lhi * 8;
      qf[c].u[0] = *(const uint4*)(qb + (size_t)qrow * rs + hd0);
      qf[c].u[1] = *(const uint4*)(qb + (size_t)qrow * rs + hd0 + 16);
    }
  }

  v8f accO[8];
  #pragma unroll
  for (int t = 0; t < 8; ++t) accO[t] = zero8();
  float rmax[8], rsum[8];
  #pragma unroll
  for (int v = 0; v < 8; ++v) { rmax[v] = -1e30f; rsum[v] = 0.0f; }

  const int kend = q0 + 128;
  for (int kblk = 0; kblk < kend; kblk += 32) {
    // ---- async stage: K rows (contiguous) + Vt chunk (contiguous) ----
    #pragma unroll
    for (int rep = 0; rep < 2; ++rep) {
      int cid = tid + rep * 256;
      int kr = cid >> 4;                 // 0..31
      int kc = (cid & 15) * 8;           // 0..120
      copy16_async(kb_ + (size_t)(kblk + kr) * rs + kc, &Ksh[kr][kc]);
      int hd  = cid >> 2;                // 0..127
      int kc2 = (cid & 3) * 8;           // 0..24
      copy16_async(vtb + (size_t)hd * S + kblk + kc2, &Vt[hd][kc2]);
    }
    async_join();
    __syncthreads();

    // ---- scores: two 16x16 tiles over this 32-key chunk ----
    v8f sc[2];
    #pragma unroll
    for (int t = 0; t < 2; ++t) {
      sc[t] = zero8();
      #pragma unroll
      for (int c = 0; c < 4; ++c) {
        F16x16 kf;
        int kn  = t * 16 + lhalf;
        int hdb = c * 32 + lhi * 8;
        kf.u[0] = *(const uint4*)&Ksh[kn][hdb];
        kf.u[1] = *(const uint4*)&Ksh[kn][hdb + 16];
        sc[t] = __builtin_amdgcn_wmma_f32_16x16x32_f16(
            false, qf[c].h, false, kf.h, (short)0, sc[t], false, false);
      }
    }

    // ---- online softmax (row m = v + 8*lhi, col = lhalf) ----
    const int rowbase = q0 + wave * 16 + 8 * lhi;
    float curmax[8];
    #pragma unroll
    for (int v = 0; v < 8; ++v) {
      int r = rowbase + v;
      float s0 = sc[0][v] * scale; if (kblk + lhalf      > r) s0 = -1e30f;
      float s1 = sc[1][v] * scale; if (kblk + 16 + lhalf > r) s1 = -1e30f;
      sc[0][v] = s0; sc[1][v] = s1;
      curmax[v] = fmaxf(s0, s1);
    }
    #pragma unroll
    for (int off = 1; off < 16; off <<= 1)
      #pragma unroll
      for (int v = 0; v < 8; ++v)
        curmax[v] = fmaxf(curmax[v], __shfl_xor(curmax[v], off, 32));

    float psum[8];
    #pragma unroll
    for (int v = 0; v < 8; ++v) {
      float nm = fmaxf(rmax[v], curmax[v]);
      float corr = __expf(rmax[v] - nm);
      rmax[v] = nm;
      float p0 = __expf(sc[0][v] - nm);
      float p1 = __expf(sc[1][v] - nm);
      psum[v] = p0 + p1;
      rsum[v] *= corr;
      #pragma unroll
      for (int t = 0; t < 8; ++t) accO[t][v] *= corr;
      Psh[wave][v + 8 * lhi][lhalf]      = (_Float16)p0;
      Psh[wave][v + 8 * lhi][16 + lhalf] = (_Float16)p1;
    }
    #pragma unroll
    for (int off = 1; off < 16; off <<= 1)
      #pragma unroll
      for (int v = 0; v < 8; ++v)
        psum[v] += __shfl_xor(psum[v], off, 32);
    #pragma unroll
    for (int v = 0; v < 8; ++v) rsum[v] += psum[v];

    // ---- P (16x32) as A-fragment, then P·V ----
    F16x16 pf;
    pf.u[0] = *(const uint4*)&Psh[wave][lhalf][lhi * 8];
    pf.u[1] = *(const uint4*)&Psh[wave][lhalf][lhi * 8 + 16];
    #pragma unroll
    for (int t = 0; t < 8; ++t) {
      F16x16 vf;
      int hn  = t * 16 + lhalf;
      int kbv = lhi * 8;
      vf.u[0] = *(const uint4*)&Vt[hn][kbv];
      vf.u[1] = *(const uint4*)&Vt[hn][kbv + 16];
      accO[t] = __builtin_amdgcn_wmma_f32_16x16x32_f16(
          false, pf.h, false, vf.h, (short)0, accO[t], false, false);
    }
    __syncthreads();
  }

  // ---- normalize + store [B,S,H] f16 ----
  #pragma unroll
  for (int v = 0; v < 8; ++v) {
    float inv = 1.0f / rsum[v];
    int srow = q0 + wave * 16 + v + 8 * lhi;
    #pragma unroll
    for (int t = 0; t < 8; ++t) {
      int col = head * HD + t * 16 + lhalf;
      out[(size_t)(b * S + srow) * H + col] = (_Float16)(accO[t][v] * inv);
    }
  }
}

// ---------------- host-side launch ----------------
extern "C" void kernel_launch(void* const* d_in, const int* in_sizes, int n_in,
                              void* d_out, int out_size, void* d_ws, size_t ws_size,
                              hipStream_t stream) {
  const int*   positions   = (const int*)d_in[0];
  const float* hidden      = (const float*)d_in[1];
  const int*   qweight_qkv = (const int*)d_in[2];
  const float* scales_qkv  = (const float*)d_in[3];
  const int*   qzeros_qkv  = (const int*)d_in[4];
  const int*   qweight_o   = (const int*)d_in[5];
  const float* scales_o    = (const float*)d_in[6];
  const int*   qzeros_o    = (const int*)d_in[7];

  char* ws = (char*)d_ws;
  size_t off = 0;
  _Float16* x_f16    = (_Float16*)(ws + off); off += (size_t)MROWS * H * 2;     // 16 MB
  _Float16* wqkvT    = (_Float16*)(ws + off); off += (size_t)H * OQKV * 2;      // 96 MB  [OQKV, H]
  _Float16* woT      = (_Float16*)(ws + off); off += (size_t)H * H * 2;         // 32 MB  [H, H]
  _Float16* qkv_f16  = (_Float16*)(ws + off); off += (size_t)MROWS * OQKV * 2;  // 48 MB
  _Float16* attn_f16 = (_Float16*)(ws + off); off += (size_t)MROWS * H * 2;     // 16 MB
  _Float16* vt_f16   = (_Float16*)(ws + off); off += (size_t)B * NH * HD * S * 2; // 16 MB

  // 1. hidden f32 -> f16
  {
    int n = MROWS * H;
    convert_f32_to_f16<<<(n + 255) / 256, 256, 0, stream>>>(hidden, x_f16, n);
  }
  // 2. dequant AWQ weights -> transposed f16
  {
    dequant_awq_T<<<(OQKV / 32) * (H / 64), 256, 0, stream>>>(
        qweight_qkv, scales_qkv, qzeros_qkv, wqkvT, H, OQKV);
    dequant_awq_T<<<(H / 32) * (H / 64), 256, 0, stream>>>(
        qweight_o, scales_o, qzeros_o, woT, H, H);
  }
  // 3. QKV GEMM: [2048,4096] x Wt[12288,4096] -> f16 [2048,12288]
  {
    dim3 grid(OQKV / 128, MROWS / 128);
    gemm_wmma<false><<<grid, 256, 0, stream>>>(x_f16, wqkvT, qkv_f16,
                                               MROWS, OQKV, H);
  }
  // 4. RoPE in place on q,k
  {
    int n = B * S * 2 * NH * (HD / 2);
    rope_kernel<<<(n + 255) / 256, 256, 0, stream>>>(positions, qkv_f16, n);
  }
  // 5. V transpose
  {
    int n = B * NH * HD * S;
    transpose_v<<<(n + 255) / 256, 256, 0, stream>>>(qkv_f16, vt_f16, n);
  }
  // 6. causal flash attention
  {
    dim3 grid(S / 128, B * NH);
    attn_kernel<<<grid, 256, 0, stream>>>(qkv_f16, vt_f16, attn_f16);
  }
  // 7. O-projection: [2048,4096] x Wt[4096,4096] -> f32 d_out
  {
    dim3 grid(H / 128, MROWS / 128);
    gemm_wmma<true><<<grid, 256, 0, stream>>>(attn_f16, woT, d_out,
                                              MROWS, H, H);
  }
}
